// ReadoutLayer_47545287967106
// MI455X (gfx1250) — matrix operand order repaired and checked by reading, not compiled, not run on hardware
//
#include <hip/hip_runtime.h>

// CDNA5 / gfx1250, wave32.
typedef __attribute__((ext_vector_type(16))) __bf16 v16bf;
typedef __attribute__((ext_vector_type(8)))  __bf16 v8bf;
typedef __attribute__((ext_vector_type(4)))  __bf16 v4bf;
typedef __attribute__((ext_vector_type(8)))  float  v8f;

#define D2        512          // 2*D (feature dim after concat)
#define XS_STRIDE 520          // 512 + 8 bf16 pad -> 260 dwords/row, avoids LDS bank conflicts

// Fused: out-partial[wgx][j] = sum over this WG's row tiles of relu(x@W^T + b)[.,j]
// blockDim = 256 (8 waves). grid = (GX, 4): blockIdx.y selects a 128-col slice.
__global__ __launch_bounds__(256) void readout_main(
    const float* __restrict__ v0, const float* __restrict__ v1,
    const float* __restrict__ W,  const float* __restrict__ b,
    float* __restrict__ partial, int mt_total)
{
  __shared__ __bf16 xs[16 * XS_STRIDE];   // 16x512 bf16 x-tile, padded

  const int tid  = threadIdx.x;
  const int wgx  = blockIdx.x;
  const int gx   = gridDim.x;
  const int wv   = tid >> 5;       // wave id 0..7
  const int lane = tid & 31;
  const int n    = lane & 15;      // column within 16-wide tile
  const int h    = lane >> 4;      // half of wave
  const int j    = blockIdx.y * 128 + wv * 16 + n;   // global output column
  const float bias = b[j];

  // ---- One-time: this wave's B fragments (W row j, all K=512) resident in VGPRs,
  //      converted f32 -> bf16. B[k][j] = W[j][k]; per-lane 16 consecutive k.
  v16bf bfrag[16];
  {
    const float* wrow = W + (size_t)j * D2;
#pragma unroll
    for (int s = 0; s < 16; ++s) {
      v16bf bb;
#pragma unroll
      for (int c = 0; c < 16; ++c)
        bb[c] = (__bf16)wrow[s * 32 + h * 16 + c];
      bfrag[s] = bb;
    }
  }

  float colsum = 0.0f;

  for (int t = wgx; t < mt_total; t += gx) {
    const int row0 = t << 4;
    __syncthreads();   // previous tile's LDS reads done before overwrite

    // ---- Stage x-tile: concat(v0,v1) rows -> bf16 LDS. 8192 elems / 256 thr.
#pragma unroll
    for (int i = 0; i < 8; ++i) {
      const int flat = ((i << 8) + tid) << 2;       // elem index, step 4
      const int r = flat >> 9;                      // row in tile
      const int k = flat & 511;                     // feature index
      const float* src = (k < 256)
          ? (v0 + (size_t)(row0 + r) * 256 + k)
          : (v1 + (size_t)(row0 + r) * 256 + (k - 256));
      const float4 f = *(const float4*)src;
      v4bf o;
      o[0] = (__bf16)f.x; o[1] = (__bf16)f.y;
      o[2] = (__bf16)f.z; o[3] = (__bf16)f.w;
      *(v4bf*)(&xs[r * XS_STRIDE + k]) = o;
    }

    // Prefetch next tile's x rows into cache hierarchy (global_prefetch_b8)
    const int tn = t + gx;
    if (tn < mt_total) {
      const size_t base = (size_t)(tn << 4) * 256 + (size_t)(tid << 2);
      __builtin_prefetch(v0 + base, 0, 1);
      __builtin_prefetch(v1 + base, 0, 1);
    }

    __syncthreads();

    // ---- 16x16 tile GEMM: K=512 in 16 steps of v_wmma_f32_16x16x32_bf16.
    // Two A-fragment buffers with overlapping live ranges (a1 loaded before
    // wmma(a0) and consumed after it) so the coalescer cannot merge them:
    // step s+1's ds_load_b128 pair overlaps WMMA s instead of waiting dscnt==0.
    v8f acc = {};
    const __bf16* xrow = &xs[n * XS_STRIDE + h * 16];
    union AU { v16bf v; v8bf p[2]; };
    AU a0, a1;
    a0.p[0] = *(const v8bf*)(xrow);
    a0.p[1] = *(const v8bf*)(xrow + 8);
#pragma unroll
    for (int s = 0; s < 16; s += 2) {
      // preload step s+1 into the *other* buffer before consuming a0
      a1.p[0] = *(const v8bf*)(xrow + (s + 1) * 32);
      a1.p[1] = *(const v8bf*)(xrow + (s + 1) * 32 + 8);
      acc = __builtin_amdgcn_wmma_f32_16x16x32_bf16(
                false, a0.v, false, bfrag[s], (short)0, acc, false, false);
      if (s + 2 < 16) {
        a0.p[0] = *(const v8bf*)(xrow + (s + 2) * 32);
        a0.p[1] = *(const v8bf*)(xrow + (s + 2) * 32 + 8);
      }
      acc = __builtin_amdgcn_wmma_f32_16x16x32_bf16(
                false, a1.v, false, bfrag[s + 1], (short)0, acc, false, false);
    }

    // ---- Fused bias + ReLU + row-sum (this lane owns 8 rows of column j)
#pragma unroll
    for (int r = 0; r < 8; ++r) {
      const float e = acc[r] + bias;
      colsum += (e > 0.0f) ? e : 0.0f;
    }
  }

  // Combine the two half-wave row groups (M 0..7 and 8..15) and emit partial.
  const float other = __shfl_xor(colsum, 16, 32);
  if (h == 0)
    partial[(size_t)wgx * D2 + j] = colsum + other;
}

// Deterministic final reduction over per-WG partials.
__global__ __launch_bounds__(256) void readout_reduce(
    const float* __restrict__ partial, float* __restrict__ out, int gx)
{
  const int j = blockIdx.x * 256 + threadIdx.x;
  if (j < D2) {
    float s = 0.0f;
    for (int i = 0; i < gx; ++i)
      s += partial[(size_t)i * D2 + j];
    out[j] = s;
  }
}

extern "C" void kernel_launch(void* const* d_in, const int* in_sizes, int n_in,
                              void* d_out, int out_size, void* d_ws, size_t ws_size,
                              hipStream_t stream) {
  const float* v0 = (const float*)d_in[0];   // [N,256] f32
  const float* v1 = (const float*)d_in[1];   // [N,256] f32
  const float* W  = (const float*)d_in[2];   // [512,512] f32
  const float* b  = (const float*)d_in[3];   // [512] f32
  float* out = (float*)d_out;                // [1,512] f32

  const int N = in_sizes[0] / 256;           // 200000
  const int mt_total = N >> 4;               // 12500 row tiles of 16

  const int GX = 625;                        // row-tile stride; 2500 WGs total
  float* partial = (float*)d_ws;             // GX*512 floats = 1.25 MB scratch

  dim3 grid(GX, 4);
  readout_main<<<grid, 256, 0, stream>>>(v0, v1, W, b, partial, mt_total);
  readout_reduce<<<2, 256, 0, stream>>>(partial, out, GX);
}
